// ScaledDotProductAttention_69630009803448
// MI455X (gfx1250) — matrix operand order repaired and checked by reading, not compiled, not run on hardware
//
#include <hip/hip_runtime.h>

typedef __attribute__((ext_vector_type(16))) _Float16 v16h;
typedef __attribute__((ext_vector_type(4)))  _Float16 v4h;
typedef __attribute__((ext_vector_type(8)))  float    v8f;

#define S_LEN  2048
#define D_KV   64
#define NHEAD  16
#define NWAVES 4
#define QTILE  16

__device__ __forceinline__ v8f wmma_f16(v16h a, v16h b, v8f c) {
    return __builtin_amdgcn_wmma_f32_16x16x32_f16(false, a, false, b, (short)0, c, false, false);
}

// ---------------- preprocessing: K -> f16 row-major ----------------
__global__ __launch_bounds__(256)
void convert_k_f16(const float* __restrict__ K, _Float16* __restrict__ Kh, long n4) {
    long i = (long)blockIdx.x * blockDim.x + threadIdx.x;
    if (i < n4) {
        float4 v = ((const float4*)K)[i];
        v4h h = { (_Float16)v.x, (_Float16)v.y, (_Float16)v.z, (_Float16)v.w };
        ((v4h*)Kh)[i] = h;
    }
}

// ---------------- preprocessing: V -> f16 transposed [bh][d][key] ----------------
__global__ __launch_bounds__(256)
void transpose_v_f16(const float* __restrict__ V, _Float16* __restrict__ Vt) {
    __shared__ _Float16 tile[64][72];
    const int bh = blockIdx.y;
    const int kb = blockIdx.x * 64;
    const int tid = threadIdx.x;
    const int c     = tid & 63;   // read: d index / write: key index
    const int rbase = tid >> 6;   // 0..3

    const float* Vp = V + ((size_t)bh * S_LEN + kb) * D_KV;
    #pragma unroll
    for (int i = 0; i < 16; ++i) {
        int key = i * 4 + rbase;
        tile[key][c] = (_Float16)Vp[(size_t)key * D_KV + c];   // coalesced along d
    }
    __syncthreads();
    _Float16* Vtp = Vt + (size_t)bh * D_KV * S_LEN + kb;
    #pragma unroll
    for (int i = 0; i < 16; ++i) {
        int d = i * 4 + rbase;
        Vtp[(size_t)d * S_LEN + c] = tile[c][d];               // coalesced along key
    }
}

// ---------------- fused attention ----------------
template <bool PRE>
__global__ __launch_bounds__(NWAVES * 32)
void sdpa_fused_kernel(const float* __restrict__ Q, const float* __restrict__ K,
                       const float* __restrict__ V,
                       const _Float16* __restrict__ Kh, const _Float16* __restrict__ Vt,
                       const unsigned char* __restrict__ mask,
                       float* __restrict__ ctx, float* __restrict__ attn)
{
    const int tid  = threadIdx.x;
    const int wave = tid >> 5;
    const int lane = tid & 31;
    const int hl   = lane >> 4;   // which 16-lane half
    const int ln   = lane & 15;

    const int bh    = blockIdx.y;
    const int b     = bh / NHEAD;
    const int qbase = blockIdx.x * (NWAVES * QTILE) + wave * QTILE;

    const float*    Qp  = Q  + (size_t)bh * S_LEN * D_KV;
    const float*    Kp  = K  + (size_t)bh * S_LEN * D_KV;
    const float*    Vp  = V  + (size_t)bh * S_LEN * D_KV;
    const _Float16* Khp = PRE ? (Kh + (size_t)bh * S_LEN * D_KV) : nullptr;
    const _Float16* Vtp = PRE ? (Vt + (size_t)bh * D_KV * S_LEN) : nullptr;
    const unsigned char* Mp = mask + (size_t)b * S_LEN * S_LEN;
    float* Cp = ctx  + (size_t)bh * S_LEN * D_KV;
    float* Ap = attn + (size_t)bh * S_LEN * S_LEN;

    // per-wave staging of probability tile (16 rows x 64 keys), padded vs bank conflicts
    __shared__ _Float16 lds_p[NWAVES][16][72];

    // ---- Q A-fragments (16x32 f16, contiguous k = 16*hl + e), softmax scale folded in ----
    v16h aq0, aq1;
    {
        const float* qr = Qp + (size_t)(qbase + ln) * D_KV;
        #pragma unroll
        for (int e = 0; e < 8; ++e) {
            aq0[e]     = (_Float16)(qr[ 0 + hl*16 + e] * 0.125f);
            aq0[8 + e] = (_Float16)(qr[ 8 + hl*16 + e] * 0.125f);
            aq1[e]     = (_Float16)(qr[32 + hl*16 + e] * 0.125f);
            aq1[8 + e] = (_Float16)(qr[40 + hl*16 + e] * 0.125f);
        }
    }

    // ================= pass 1: per-lane-column online (max, sumexp) =================
    float m[8], l[8];
    #pragma unroll
    for (int r = 0; r < 8; ++r) { m[r] = -3.0e38f; l[r] = 0.0f; }

    for (int kt = 0; kt < S_LEN / 16; ++kt) {
        v16h bk0, bk1;
        if (PRE) {
            const _Float16* kr = Khp + (size_t)(kt * 16 + ln) * D_KV;
            bk0 = *(const v16h*)(kr +  0 + hl * 16);
            bk1 = *(const v16h*)(kr + 32 + hl * 16);
        } else {
            const float* kr = Kp + (size_t)(kt * 16 + ln) * D_KV;
            #pragma unroll
            for (int e = 0; e < 16; ++e) {
                bk0[e] = (_Float16)kr[ 0 + hl*16 + e];
                bk1[e] = (_Float16)kr[32 + hl*16 + e];
            }
        }
        v8f acc = {};
        acc = wmma_f16(aq0, bk0, acc);
        acc = wmma_f16(aq1, bk1, acc);

        const unsigned char* mrow = Mp + (size_t)(qbase + 8*hl) * S_LEN + kt * 16 + ln;
        #pragma unroll
        for (int r = 0; r < 8; ++r) {
            float s  = mrow[(size_t)r * S_LEN] ? -1.0e9f : acc[r];
            float mx = fmaxf(m[r], s);
            float e  = __expf(fminf(m[r], s) - mx);   // single exp per element
            l[r] = (s <= m[r]) ? (l[r] + e) : (l[r] * e + 1.0f);
            m[r] = mx;
        }
    }

    // merge (m,l) across the 16 lanes of each half (associative softmax combine)
    #pragma unroll
    for (int off = 1; off < 16; off <<= 1) {
        #pragma unroll
        for (int r = 0; r < 8; ++r) {
            float om = __shfl_xor(m[r], off, 32);
            float ol = __shfl_xor(l[r], off, 32);
            float mx = fmaxf(m[r], om);
            l[r] = l[r] * __expf(m[r] - mx) + ol * __expf(om - mx);
            m[r] = mx;
        }
    }
    float rinv[8];
    #pragma unroll
    for (int r = 0; r < 8; ++r) rinv[r] = 1.0f / l[r];

    // ================= pass 2: recompute, write attn, accumulate context =================
    v8f O[4] = {};

    for (int kc = 0; kc < S_LEN / 64; ++kc) {
        // recompute 4 score tiles, stage normalized probs into LDS (f16)
        #pragma unroll
        for (int t = 0; t < 4; ++t) {
            const int kt = kc * 4 + t;
            v16h bk0, bk1;
            if (PRE) {
                const _Float16* kr = Khp + (size_t)(kt * 16 + ln) * D_KV;
                bk0 = *(const v16h*)(kr +  0 + hl * 16);
                bk1 = *(const v16h*)(kr + 32 + hl * 16);
            } else {
                const float* kr = Kp + (size_t)(kt * 16 + ln) * D_KV;
                #pragma unroll
                for (int e = 0; e < 16; ++e) {
                    bk0[e] = (_Float16)kr[ 0 + hl*16 + e];
                    bk1[e] = (_Float16)kr[32 + hl*16 + e];
                }
            }
            v8f acc = {};
            acc = wmma_f16(aq0, bk0, acc);
            acc = wmma_f16(aq1, bk1, acc);

            const unsigned char* mrow = Mp + (size_t)(qbase + 8*hl) * S_LEN + kt * 16 + ln;
            #pragma unroll
            for (int r = 0; r < 8; ++r) {
                float s = mrow[(size_t)r * S_LEN] ? -1.0e9f : acc[r];
                float p = __expf(s - m[r]) * rinv[r];
                lds_p[wave][r + 8*hl][t * 16 + ln] = (_Float16)p;
            }
        }

        // coalesced attn writeback: lane covers (row = lane/2, 32 cols)
        {
            const int row = lane >> 1;
            const int cb  = (lane & 1) * 32;
            float* arow = Ap + (size_t)(qbase + row) * S_LEN + kc * 64 + cb;
            #pragma unroll
            for (int c4 = 0; c4 < 8; ++c4) {
                float4 o;
                o.x = (float)lds_p[wave][row][cb + c4*4 + 0];
                o.y = (float)lds_p[wave][row][cb + c4*4 + 1];
                o.z = (float)lds_p[wave][row][cb + c4*4 + 2];
                o.w = (float)lds_p[wave][row][cb + c4*4 + 3];
                *(float4*)(arow + c4 * 4) = o;
            }
        }

        // PV: A-fragments straight from staged f16 LDS, B-fragments = V columns
        #pragma unroll
        for (int sub = 0; sub < 2; ++sub) {
            v16h ap;
            #pragma unroll
            for (int e = 0; e < 8; ++e) {
                ap[e]     = lds_p[wave][ln][sub*32 +     hl*16 + e];
                ap[8 + e] = lds_p[wave][ln][sub*32 + 8 + hl*16 + e];
            }
            #pragma unroll
            for (int t = 0; t < 4; ++t) {
                v16h bv;
                if (PRE) {
                    // Vt is [d][key]: one contiguous 32B load per fragment
                    const _Float16* vr = Vtp + (size_t)(t*16 + ln) * S_LEN
                                             + kc*64 + sub*32 + hl*16;
                    bv = *(const v16h*)vr;
                } else {
                    const float* vcol = Vp + (size_t)(kc*64 + sub*32 + hl*16) * D_KV + t*16 + ln;
                    #pragma unroll
                    for (int e = 0; e < 16; ++e)
                        bv[e] = (_Float16)vcol[(size_t)e * D_KV];
                }
                O[t] = wmma_f16(ap, bv, O[t]);
            }
        }
    }

    // context writeback (C-layout: row = r + 8*hl, col = t*16 + ln)
    #pragma unroll
    for (int t = 0; t < 4; ++t) {
        #pragma unroll
        for (int r = 0; r < 8; ++r) {
            Cp[(size_t)(qbase + r + 8*hl) * D_KV + t * 16 + ln] = O[t][r];
        }
    }
}

extern "C" void kernel_launch(void* const* d_in, const int* in_sizes, int n_in,
                              void* d_out, int out_size, void* d_ws, size_t ws_size,
                              hipStream_t stream) {
    (void)in_sizes; (void)n_in; (void)out_size;
    const float* Q = (const float*)d_in[0];
    const float* K = (const float*)d_in[1];
    const float* V = (const float*)d_in[2];
    const unsigned char* mask = (const unsigned char*)d_in[3];

    const int B = 2;
    const size_t elems = (size_t)B * NHEAD * S_LEN * D_KV;         // 4.19M
    float* ctx  = (float*)d_out;                                   // [B,H,S,D]
    float* attn = ctx + elems;                                     // [B,H,S,S]

    dim3 grid(S_LEN / (NWAVES * QTILE), B * NHEAD);
    const size_t need = elems * 2 * sizeof(_Float16);              // Kh + Vt = 16.8MB

    if (ws_size >= need) {
        _Float16* Kh = (_Float16*)d_ws;
        _Float16* Vt = Kh + elems;
        convert_k_f16<<<(int)(elems / 4 / 256), 256, 0, stream>>>(K, Kh, (long)(elems / 4));
        transpose_v_f16<<<dim3(S_LEN / 64, B * NHEAD), 256, 0, stream>>>(V, Vt);
        sdpa_fused_kernel<true><<<grid, NWAVES * 32, 0, stream>>>(
            Q, K, V, Kh, Vt, mask, ctx, attn);
    } else {
        sdpa_fused_kernel<false><<<grid, NWAVES * 32, 0, stream>>>(
            Q, K, V, nullptr, nullptr, mask, ctx, attn);
    }
}
